// DIAM_19842748907619
// MI455X (gfx1250) — matrix operand order repaired and checked by reading, not compiled, not run on hardware
//
#include <hip/hip_runtime.h>

// ---------------------------------------------------------------------------
// Problem constants (match reference): B=8, CQ=128, CV=64, H=W=192, K=4
// ---------------------------------------------------------------------------
#define BB   8
#define HH   192
#define WW   192
#define HWP  (HH * WW)          // 36864 pixels per image
#define CQN  128
#define CVN  64
#define KTOT 1152               // 9 taps * 128 cin

typedef unsigned int u32;
typedef __attribute__((ext_vector_type(16))) __bf16 v16bf;
typedef __attribute__((ext_vector_type(8)))  float  v8f;
typedef int v4i __attribute__((vector_size(16)));

union FragBF { uint4 u[2]; v16bf v; };

// ---------------------------------------------------------------------------
// CDNA5 async global->LDS staging (guarded; falls back to sync copy)
// ---------------------------------------------------------------------------
#if defined(__has_builtin)
#if __has_builtin(__builtin_amdgcn_global_load_async_to_lds_b128)
#define HAVE_ASYNC_B128 1
#endif
#if __has_builtin(__builtin_amdgcn_s_wait_asynccnt)
#define HAVE_WAIT_ASYNC 1
#endif
#endif

__device__ __forceinline__ void async_copy16(const void* g, void* l) {
#ifdef HAVE_ASYNC_B128
  __builtin_amdgcn_global_load_async_to_lds_b128(
      (__attribute__((address_space(1))) v4i*)(const_cast<void*>(g)),
      (__attribute__((address_space(3))) v4i*)l,
      0, 0);
#else
  *(uint4*)l = *(const uint4*)g;
#endif
}

__device__ __forceinline__ void wait_async() {
#ifdef HAVE_ASYNC_B128
#ifdef HAVE_WAIT_ASYNC
  __builtin_amdgcn_s_wait_asynccnt(0);
#else
  asm volatile("s_wait_asynccnt 0" ::: "memory");
#endif
#endif
}

// f32 -> bf16 (round-to-nearest-even), raw bits
__device__ __forceinline__ unsigned short f2bf(float f) {
  union { float f; u32 u; } v; v.f = f;
  u32 u = v.u;
  u32 r = (u + 0x7FFFu + ((u >> 16) & 1u)) >> 16;
  return (unsigned short)r;
}

__device__ __forceinline__ v8f wmma_bf16(v16bf a, v16bf b, v8f c) {
  return __builtin_amdgcn_wmma_f32_16x16x32_bf16(
      false, a, false, b, (short)0, c, false, false);
}

// ---------------------------------------------------------------------------
// Fragment loaders from LDS tiles laid out as rows of 16 uint pairs,
// row stride = 20 uints (80B, 16B aligned, bank-spread).
//
// A (16x32 bf16, ISA 7.12.2): lanes 0-15 hold pairs {0..3, 8..11},
//                             lanes 16-31 hold pairs {4..7, 12..15}.
// B (32x16 bf16): lanes 0-15 hold pairs {0..3, 8..11} for N=lane,
//                 lanes 16-31 hold pairs {4..7, 12..15} for N=lane-16.
// ---------------------------------------------------------------------------
__device__ __forceinline__ v16bf load_a_frag(const u32* row, int half) {
  FragBF f;
  f.u[0] = ((const uint4*)row)[half];
  f.u[1] = ((const uint4*)row)[half + 2];
  return f.v;
}
__device__ __forceinline__ v16bf load_b_frag(const u32* row, int half) {
  FragBF f;
  f.u[0] = ((const uint4*)row)[half];
  f.u[1] = ((const uint4*)row)[half + 2];
  return f.v;
}

// ---------------------------------------------------------------------------
// Kernel 0: pack conv weights to bf16 GEMM A-matrices.
//  Apack  [128][1152]: rows 0..63 = ow1, rows 64..127 = ww1; k = tap*128+cin
//  A2pack [16][1152] : rows 0..7 = ow2 (k = tap*128+ch, ch<64),
//                      rows 8..11 = ww2 (k = tap*128+64+ch2), rest zero.
// ---------------------------------------------------------------------------
__global__ __launch_bounds__(256) void pack_weights_kernel(
    const float* __restrict__ ow1, const float* __restrict__ ww1,
    const float* __restrict__ ow2, const float* __restrict__ ww2,
    unsigned short* __restrict__ Apack, unsigned short* __restrict__ A2pack) {
  const int total1 = 128 * KTOT;
  const int total2 = 16 * KTOT;
  int gid = blockIdx.x * 256 + threadIdx.x;
  if (gid < total1) {
    int m = gid / KTOT, k = gid % KTOT;
    int tap = k >> 7, cin = k & 127;
    float val = (m < 64) ? ow1[(m * 128 + cin) * 9 + tap]
                         : ww1[((m - 64) * 128 + cin) * 9 + tap];
    Apack[gid] = f2bf(val);
  } else if (gid < total1 + total2) {
    int g2 = gid - total1;
    int m = g2 / KTOT, k = g2 % KTOT;
    int tap = k >> 7, ch = k & 127;
    float val = 0.0f;
    if (m < 8 && ch < 64)              val = ow2[(m * 64 + ch) * 9 + tap];
    else if (m >= 8 && m < 12 && ch >= 64)
                                       val = ww2[((m - 8) * 64 + (ch - 64)) * 9 + tap];
    A2pack[g2] = f2bf(val);
  }
}

// ---------------------------------------------------------------------------
// Kernel 1: conv1 (both branches fused) + bias + GELU -> hidden bf16 [B,128,H,W]
// Implicit GEMM: M=128, K=1152, N=pixels. Block: 256 threads = 8 waves as
// 4 M-groups x 2 N-groups; each wave owns a 32x32 output = 2x2 WMMA tiles
// (4 accumulators, 4 WMMAs per staged K-step, fragments reused).
// Block N tile = 64 pixels along W in one row. 36 K-steps of 32.
// ---------------------------------------------------------------------------
__global__ __launch_bounds__(256) void conv1_kernel(
    const float* __restrict__ q, const unsigned short* __restrict__ Apack,
    const float* __restrict__ ob1, const float* __restrict__ wb1,
    unsigned short* __restrict__ hidden) {
  __shared__ __align__(16) u32 lA[128 * 20];   // 128 rows x 32 bf16 (+pad)
  __shared__ __align__(16) u32 lB[64 * 20];    // 64 pixels x 32 bf16 (+pad)

  const int tid = threadIdx.x;
  const int blk = blockIdx.x;
  const int b  = blk / (HH * 3);
  const int y  = (blk / 3) % HH;
  const int x0 = (blk % 3) * 64;

  const int lane = tid & 31, half = lane >> 4, lr = lane & 15;
  const int w = tid >> 5, wm = w >> 1, wn = w & 1;

  v8f acc[2][2] = {{{}, {}}, {{}, {}}};

  for (int s = 0; s < 36; ++s) {
    const int t = s >> 2, cc = s & 3;
    const int kb = t * 128 + cc * 32;
    const int dy = t / 3 - 1, dx = t % 3 - 1;

    __syncthreads();

    // --- stage A: 128x32 bf16 weight chunk, async global->LDS, 2x16B/thread
#pragma unroll
    for (int i = 0; i < 2; ++i) {
      const int p = tid + 256 * i;
      const int m = p >> 2, aq = p & 3;
      async_copy16((const char*)Apack + (size_t)m * (KTOT * 2) + kb * 2 + aq * 16,
                   (char*)lA + m * 80 + aq * 16);
    }

    // --- stage B: 32 cin x 64 px of q, f32->bf16, branchless halo (clamp+select)
    {
      const int yy = y + dy;
      const int yc = min(max(yy, 0), HH - 1);
      const bool yok = (yy >= 0) && (yy < HH);
      unsigned short* sB = (unsigned short*)lB;
#pragma unroll
      for (int i = 0; i < 4; ++i) {
        const int e = tid + 256 * i;
        const int n = e & 63, k0 = e >> 6;          // k0 in 0..15
        const int xx = x0 + n + dx;
        const int xc = min(max(xx, 0), WW - 1);
        const bool ok = yok && (xx >= 0) && (xx < WW);
        const int cin0 = cc * 32 + k0;
        const float* qp = q + (((size_t)b * CQN + cin0) * HH + yc) * WW + xc;
        const float f0 = qp[0];
        const float f1 = qp[(size_t)16 * HWP];
        sB[n * 40 + k0]      = ok ? f2bf(f0) : (unsigned short)0;
        sB[n * 40 + k0 + 16] = ok ? f2bf(f1) : (unsigned short)0;
      }
    }

    wait_async();
    __syncthreads();

    const u32* rowA0 = lA + (wm * 32 + lr) * 20;
    const u32* rowA1 = rowA0 + 16 * 20;
    const u32* rowB0 = lB + (wn * 32 + lr) * 20;
    const u32* rowB1 = rowB0 + 16 * 20;
    const v16bf a0 = load_a_frag(rowA0, half);
    const v16bf a1 = load_a_frag(rowA1, half);
    const v16bf b0 = load_b_frag(rowB0, half);
    const v16bf b1 = load_b_frag(rowB1, half);
    acc[0][0] = wmma_bf16(a0, b0, acc[0][0]);
    acc[0][1] = wmma_bf16(a0, b1, acc[0][1]);
    acc[1][0] = wmma_bf16(a1, b0, acc[1][0]);
    acc[1][1] = wmma_bf16(a1, b1, acc[1][1]);
  }

  // epilogue: bias + exact GELU, store bf16 hidden
#pragma unroll
  for (int i = 0; i < 2; ++i) {
#pragma unroll
    for (int j = 0; j < 2; ++j) {
      const int px = x0 + wn * 32 + j * 16 + lr;
#pragma unroll
      for (int r = 0; r < 8; ++r) {
        const int ch = wm * 32 + i * 16 + half * 8 + r;
        const float bias = (ch < 64) ? ob1[ch] : wb1[ch - 64];
        const float xv = acc[i][j][r] + bias;
        const float g = 0.5f * xv * (1.0f + erff(xv * 0.70710678118654752f));
        hidden[(((size_t)b * CQN + ch) * HH + y) * WW + px] = f2bf(g);
      }
    }
  }
}

// ---------------------------------------------------------------------------
// Kernel 2: conv2 (offsets 8ch + logits 4ch, padded M=16) from bf16 hidden.
// Block: 128 threads = 4 waves, each wave one 16-pixel N tile (64 px/block).
// Writes P fp32 [B,12,H,W] (ch 0..7 offsets, 8..11 logits).
// ---------------------------------------------------------------------------
__global__ __launch_bounds__(128) void conv2_kernel(
    const unsigned short* __restrict__ hidden,
    const unsigned short* __restrict__ A2pack,
    const float* __restrict__ ob2, const float* __restrict__ wb2,
    float* __restrict__ P) {
  __shared__ __align__(16) u32 lA[16 * 20];
  __shared__ __align__(16) u32 lB[64 * 20];

  const int tid = threadIdx.x;
  const int blk = blockIdx.x;
  const int b  = blk / (HH * 3);
  const int y  = (blk / 3) % HH;
  const int x0 = (blk % 3) * 64;

  const int lane = tid & 31, half = lane >> 4, lr = lane & 15;
  const int w = tid >> 5;
  const int am = tid >> 2, aq = tid & 3;

  v8f acc = {};

  for (int s = 0; s < 36; ++s) {
    const int t = s >> 2, cc = s & 3;
    const int kb = t * 128 + cc * 32;
    const int dy = t / 3 - 1, dx = t % 3 - 1;

    __syncthreads();

    if (tid < 64)
      async_copy16((const char*)A2pack + (size_t)am * (KTOT * 2) + kb * 2 + aq * 16,
                   (char*)lA + am * 80 + aq * 16);

    {
      const int yy = y + dy;
      const int yc = min(max(yy, 0), HH - 1);
      const bool yok = (yy >= 0) && (yy < HH);
      unsigned short* sB = (unsigned short*)lB;
#pragma unroll
      for (int i = 0; i < 16; ++i) {
        const int e = tid + 128 * i;
        const int n = e & 63, k = e >> 6;
        const int xx = x0 + n + dx;
        const int xc = min(max(xx, 0), WW - 1);
        const bool ok = yok && (xx >= 0) && (xx < WW);
        const int ch = cc * 32 + k;
        const unsigned short hv =
            hidden[(((size_t)b * CQN + ch) * HH + yc) * WW + xc];
        sB[n * 40 + k] = ok ? hv : (unsigned short)0;
      }
    }

    if (tid < 64) wait_async();
    __syncthreads();

    const u32* rowA = lA + lr * 20;
    const u32* rowB = lB + (w * 16 + lr) * 20;
    const v16bf af = load_a_frag(rowA, half);
    const v16bf bf = load_b_frag(rowB, half);
    acc = wmma_bf16(af, bf, acc);
  }

  const int px = x0 + w * 16 + lr;
#pragma unroll
  for (int r = 0; r < 8; ++r) {
    const int m = half * 8 + r;
    if (m < 12) {
      const float bias = (m < 8) ? ob2[m] : wb2[m - 8];
      P[(((size_t)b * 12 + m) * HWP) + y * WW + px] = acc[r] + bias;
    }
  }
}

// ---------------------------------------------------------------------------
// Kernel 3: per-pixel softmax over K=4, bilinear gather from v, weighted sum.
// One thread per pixel; precompute 16 (index, weight) pairs, loop 64 channels.
// ---------------------------------------------------------------------------
__global__ __launch_bounds__(256) void sample_kernel(
    const float* __restrict__ v, const float* __restrict__ P,
    float* __restrict__ out) {
  const int p = blockIdx.x * 256 + threadIdx.x;
  if (p >= BB * HWP) return;
  const int b = p / HWP;
  const int r = p - b * HWP;
  const int y = r / WW;
  const int x = r % WW;

  float offs[8], lg[4];
#pragma unroll
  for (int j = 0; j < 8; ++j) offs[j] = P[((size_t)b * 12 + j) * HWP + r];
#pragma unroll
  for (int j = 0; j < 4; ++j) lg[j] = P[((size_t)b * 12 + 8 + j) * HWP + r];

  float mx = fmaxf(fmaxf(lg[0], lg[1]), fmaxf(lg[2], lg[3]));
  float e0 = __expf(lg[0] - mx), e1 = __expf(lg[1] - mx);
  float e2 = __expf(lg[2] - mx), e3 = __expf(lg[3] - mx);
  const float inv = 1.0f / (e0 + e1 + e2 + e3);
  float wk[4] = {e0 * inv, e1 * inv, e2 * inv, e3 * inv};

  int   idx[16];
  float wgt[16];
#pragma unroll
  for (int k = 0; k < 4; ++k) {
    const float fx = (float)x + offs[2 * k]     * 0.5f * (WW - 1);
    const float fy = (float)y + offs[2 * k + 1] * 0.5f * (HH - 1);
    const float x0f = floorf(fx), y0f = floorf(fy);
    const float wx1 = fx - x0f, wx0 = 1.0f - wx1;
    const float wy1 = fy - y0f, wy0 = 1.0f - wy1;
#pragma unroll
    for (int c = 0; c < 4; ++c) {
      const int dx2 = c & 1, dy2 = c >> 1;
      const float cx = x0f + dx2, cy = y0f + dy2;
      const bool valid = (cx >= 0.0f) && (cx <= (float)(WW - 1)) &&
                         (cy >= 0.0f) && (cy <= (float)(HH - 1));
      int xi = (int)fminf(fmaxf(cx, 0.0f), (float)(WW - 1));
      int yi = (int)fminf(fmaxf(cy, 0.0f), (float)(HH - 1));
      idx[k * 4 + c] = yi * WW + xi;
      wgt[k * 4 + c] = wk[k] * (dx2 ? wx1 : wx0) * (dy2 ? wy1 : wy0) *
                       (valid ? 1.0f : 0.0f);
    }
  }

  const float* vb   = v   + (size_t)b * CVN * HWP;
  float*       outb = out + (size_t)b * CVN * HWP + r;
  for (int c = 0; c < CVN; ++c) {
    const float* vc = vb + (size_t)c * HWP;
    float a = 0.0f;
#pragma unroll
    for (int i = 0; i < 16; ++i) a += wgt[i] * vc[idx[i]];
    outb[(size_t)c * HWP] = a;
  }
}

// ---------------------------------------------------------------------------
// Host launcher.  Inputs (setup_inputs order):
//  0 q, 1 v, 2 ow1, 3 ob1, 4 ow2, 5 ob2, 6 ww1, 7 wb1, 8 ww2, 9 wb2
// Workspace layout (bytes):
//  [0, 294912)            Apack   bf16 [128][1152]
//  [294912, 331776)       A2pack  bf16 [16][1152]
//  [331776, 75829248)     hidden  bf16 [B,128,H,W]
//  [75829248, ...)        P       f32  [B,12,H,W]
// ---------------------------------------------------------------------------
extern "C" void kernel_launch(void* const* d_in, const int* in_sizes, int n_in,
                              void* d_out, int out_size, void* d_ws,
                              size_t ws_size, hipStream_t stream) {
  const float* q   = (const float*)d_in[0];
  const float* v   = (const float*)d_in[1];
  const float* ow1 = (const float*)d_in[2];
  const float* ob1 = (const float*)d_in[3];
  const float* ow2 = (const float*)d_in[4];
  const float* ob2 = (const float*)d_in[5];
  const float* ww1 = (const float*)d_in[6];
  const float* wb1 = (const float*)d_in[7];
  const float* ww2 = (const float*)d_in[8];
  const float* wb2 = (const float*)d_in[9];
  float* out = (float*)d_out;

  char* ws = (char*)d_ws;
  unsigned short* Apack  = (unsigned short*)(ws);
  unsigned short* A2pack = (unsigned short*)(ws + 294912);
  unsigned short* hidden = (unsigned short*)(ws + 331776);
  float*          P      = (float*)(ws + 331776 + (size_t)75497472);

  // 1) pack weights to bf16 GEMM layout (165888 elements)
  pack_weights_kernel<<<648, 256, 0, stream>>>(ow1, ww1, ow2, ww2, Apack, A2pack);

  // 2) conv1 (fused both branches) + GELU -> hidden
  conv1_kernel<<<BB * HH * 3, 256, 0, stream>>>(q, Apack, ob1, wb1, hidden);

  // 3) conv2 (offsets + logits) -> P
  conv2_kernel<<<BB * HH * 3, 128, 0, stream>>>(hidden, A2pack, ob2, wb2, P);

  // 4) softmax + bilinear sample + weighted sum -> out
  sample_kernel<<<(BB * HWP + 255) / 256, 256, 0, stream>>>(v, P, out);
}
